// PointNet2Seg_7301444403790
// MI455X (gfx1250) — matrix operand order repaired and checked by reading, not compiled, not run on hardware
//
#include <hip/hip_runtime.h>

// =====================================================================
// PointNet++ segmentation forward for MI455X (gfx1250, wave32, WMMA).
//
// All 18 dense layers run through v_wmma_f32_16x16x32_f16 (f16 A/B,
// f32 accumulate, fused bias+ReLU). The GEMM is register-blocked 4x in
// N: one wave holds a single A fragment and 4 accumulators (16x64
// output tile). Per K step, all 10 b128 loads (A + 4 B fragments) are
// issued before any WMMA so the loadcnt waits become partial and the
// matrix pipe overlaps the fetch stream.
// ~38 GFLOPs total, ~35 MB working set -> fully L2 resident.
// =====================================================================

typedef __attribute__((ext_vector_type(16))) _Float16 v16h;
typedef __attribute__((ext_vector_type(8)))  float    v8f;
typedef _Float16 f16;

#define BB    8
#define NPTS  2048
#define NFEAT 9
#define SA1S  1024
#define SA2S  256
#define KNN   64

// ---------------------------------------------------------------------
// Weight prep: fp32 W[cin][cout] -> f16 Wt[coutP][cinP] (transposed,
// zero padded), bias -> f32 padded.
// ---------------------------------------------------------------------
__global__ void k_prep_weights(const float* __restrict__ W,
                               const float* __restrict__ bias,
                               f16* __restrict__ Wt, float* __restrict__ bp,
                               int cin, int cout, int cinP, int coutP)
{
    int tid = blockIdx.x * blockDim.x + threadIdx.x;
    int total = coutP * cinP;
    if (tid >= total) return;
    int n = tid / cinP, k = tid % cinP;
    f16 v = (f16)0.f;
    if (n < cout && k < cin) v = (f16)W[(size_t)k * cout + n];
    Wt[tid] = v;
    if (k == 0) bp[n] = (n < cout) ? bias[n] : 0.f;
}

// ---------------------------------------------------------------------
// WMMA GEMM: out[M,Np] = relu(A[M,Kp] * Wt^T + bias)
//   A  : f16 row-major, Kp multiple of 32, rows zero-padded in K
//   Wt : f16 [Np][Kp] (weights pre-transposed -> contiguous K per column)
// One wave computes a 16 x (16*NB) tile: 1 A fragment -> NB WMMAs per
// K step. Loads are grouped ahead of the WMMAs for overlap.
// ---------------------------------------------------------------------
template <int NB>
__global__ void k_gemm(const f16* __restrict__ A, const f16* __restrict__ Wt,
                       const float* __restrict__ bias, void* __restrict__ out,
                       int M, int Kp, int Np, int relu, int out_f32)
{
    int wave  = threadIdx.x >> 5;
    int lane  = threadIdx.x & 31;
    int mtile = blockIdx.x * (blockDim.x >> 5) + wave;
    if (mtile * 16 >= M) return;              // wave-uniform: EXEC stays full
    int nbase = blockIdx.y * 16 * NB;
    int hi = lane >> 4, mr = lane & 15;

    const f16* arow = A + (size_t)(mtile * 16 + mr) * Kp + hi * 16;
    const f16* brow[NB];
#pragma unroll
    for (int t = 0; t < NB; t++)
        brow[t] = Wt + (size_t)(nbase + 16 * t + mr) * Kp + hi * 16;

    union CU { v8f v; float f[8]; } acc[NB];
#pragma unroll
    for (int t = 0; t < NB; t++)
#pragma unroll
        for (int r = 0; r < 8; r++) acc[t].f[r] = 0.f;

    union AU { v16h v; f16 h[16]; };

    for (int k0 = 0; k0 < Kp; k0 += 32) {
        AU a, b[NB];
        __builtin_prefetch((const void*)(arow + k0 + 128), 0, 1);
        // ---- issue ALL loads first: A fragment + NB B fragments ----
#pragma unroll
        for (int j = 0; j < 8; j++) {          // A 16x32 f16, contiguous K order (same as B)
            a.h[j]     = arow[k0 + j];         // K = hi*16 + j
            a.h[j + 8] = arow[k0 + 8 + j];     // K = hi*16 + 8 + j
        }
#pragma unroll
        for (int t = 0; t < NB; t++)
#pragma unroll
            for (int j = 0; j < 16; j++)       // B 32x16: col = lane&15, K = hi*16+j
                b[t].h[j] = brow[t][k0 + j];
        // ---- then the NB WMMAs (partial loadcnt waits, overlap) ----
#pragma unroll
        for (int t = 0; t < NB; t++)
            acc[t].v = __builtin_amdgcn_wmma_f32_16x16x32_f16(
                false, a.v, false, b[t].v, (short)0, acc[t].v, false, false);
    }

#pragma unroll
    for (int t = 0; t < NB; t++) {
        int nc = nbase + 16 * t + mr;
        float bs = bias[nc];
#pragma unroll
        for (int r = 0; r < 8; r++) {          // C/D: lanes0-15 M=r, lanes16-31 M=r+8
            float v = acc[t].f[r] + bs;
            if (relu) v = fmaxf(v, 0.f);
            size_t o = (size_t)(mtile * 16 + hi * 8 + r) * Np + nc;
            if (out_f32) ((float*)out)[o] = v;
            else         ((f16*)out)[o]   = (f16)v;
        }
    }
}

// ---------------------------------------------------------------------
// Farthest point sampling, one block (1024 thr) per cloud, start idx 0.
// First-index tie-break matches jnp.argmax.
// ---------------------------------------------------------------------
__global__ void k_fps(const float* __restrict__ pos, int n, int S,
                      int* __restrict__ idx_out)
{
    __shared__ float px[NPTS], py[NPTS], pz[NPTS], mind[NPTS];
    __shared__ float rv[1024];
    __shared__ int   ri[1024];
    __shared__ float cx, cy, cz;
    int b = blockIdx.x;
    const float* P = pos + (size_t)b * n * 3;
    int t = threadIdx.x, T = blockDim.x;
    for (int i = t; i < n; i += T) {
        px[i] = P[i * 3]; py[i] = P[i * 3 + 1]; pz[i] = P[i * 3 + 2];
    }
    __syncthreads();
    for (int i = t; i < n; i += T) {
        float dx = px[i] - px[0], dy = py[i] - py[0], dz = pz[i] - pz[0];
        mind[i] = dx * dx + dy * dy + dz * dz;
    }
    if (t == 0) idx_out[(size_t)b * S] = 0;
    __syncthreads();
    for (int s = 1; s < S; s++) {
        float bv = -1.f; int bi = 0;
        for (int i = t; i < n; i += T) {       // i increasing -> keeps first on ties
            float v = mind[i];
            if (v > bv) { bv = v; bi = i; }
        }
        rv[t] = bv; ri[t] = bi;
        __syncthreads();
        for (int off = T >> 1; off > 0; off >>= 1) {
            if (t < off) {
                float v2 = rv[t + off]; int i2 = ri[t + off];
                if (v2 > rv[t] || (v2 == rv[t] && i2 < ri[t])) { rv[t] = v2; ri[t] = i2; }
            }
            __syncthreads();
        }
        if (t == 0) {
            int nxt = ri[0];
            idx_out[(size_t)b * S + s] = nxt;
            cx = px[nxt]; cy = py[nxt]; cz = pz[nxt];
        }
        __syncthreads();
        for (int i = t; i < n; i += T) {
            float dx = px[i] - cx, dy = py[i] - cy, dz = pz[i] - cz;
            float d = dx * dx + dy * dy + dz * dz;
            if (d < mind[i]) mind[i] = d;
        }
        __syncthreads();
    }
}

// ---------------------------------------------------------------------
// Ball query: first K in-radius neighbors by index (matches argsort trick).
// Also gathers center coordinates.
// ---------------------------------------------------------------------
__global__ void k_ballquery(const float* __restrict__ pos, int n,
                            const int* __restrict__ fidx, int S, float r2,
                            float* __restrict__ ctr_out, int* __restrict__ nbr,
                            int* __restrict__ cnt)
{
    int g = blockIdx.x * blockDim.x + threadIdx.x;
    if (g >= BB * S) return;
    int b = g / S;
    const float* P = pos + (size_t)b * n * 3;
    int ci = fidx[g];
    float cx = P[ci * 3], cy = P[ci * 3 + 1], cz = P[ci * 3 + 2];
    ctr_out[(size_t)g * 3] = cx; ctr_out[(size_t)g * 3 + 1] = cy; ctr_out[(size_t)g * 3 + 2] = cz;
    int* nb = nbr + (size_t)g * KNN;
    int c = 0;
    for (int j = 0; j < n && c < KNN; j++) {
        float dx = P[j * 3] - cx, dy = P[j * 3 + 1] - cy, dz = P[j * 3 + 2] - cz;
        if (dx * dx + dy * dy + dz * dz <= r2) nb[c++] = j;
    }
    int c0 = (c > 0) ? c : 1;
    if (c == 0) nb[0] = ci;                   // center always in its own ball anyway
    for (int k = c0; k < KNN; k++) nb[k] = nb[0];
    cnt[g] = c0;
}

// ---------------------------------------------------------------------
// Gather grouped features (per cloud b) into padded f16 A matrices.
// ---------------------------------------------------------------------
__global__ void k_gather_sa1(const float* __restrict__ x, const float* __restrict__ pos,
                             const float* __restrict__ ctr, const int* __restrict__ nbr,
                             f16* __restrict__ A, int b)
{
    int row = blockIdx.x * blockDim.x + threadIdx.x;
    if (row >= SA1S * KNN) return;
    int s = row / KNN;
    int j = nbr[((size_t)b * SA1S + s) * KNN + (row % KNN)];
    const float* xr = x   + ((size_t)b * NPTS + j) * NFEAT;
    const float* pr = pos + ((size_t)b * NPTS + j) * 3;
    const float* cr = ctr + ((size_t)b * SA1S + s) * 3;
    f16* a = A + (size_t)row * 32;
    for (int c = 0; c < NFEAT; c++) a[c] = (f16)xr[c];
    a[9]  = (f16)(pr[0] - cr[0]);
    a[10] = (f16)(pr[1] - cr[1]);
    a[11] = (f16)(pr[2] - cr[2]);
    for (int c = 12; c < 32; c++) a[c] = (f16)0.f;
}

__global__ void k_gather_sa2(const f16* __restrict__ x1, const float* __restrict__ pos1,
                             const float* __restrict__ ctr, const int* __restrict__ nbr,
                             f16* __restrict__ A, int b)
{
    int row = blockIdx.x * blockDim.x + threadIdx.x;
    if (row >= SA2S * KNN) return;
    int s = row / KNN;
    int j = nbr[((size_t)b * SA2S + s) * KNN + (row % KNN)];
    const f16*   xr = x1   + ((size_t)b * SA1S + j) * 128;
    const float* pr = pos1 + ((size_t)b * SA1S + j) * 3;
    const float* cr = ctr  + ((size_t)b * SA2S + s) * 3;
    f16* a = A + (size_t)row * 160;
    for (int c = 0; c < 128; c++) a[c] = xr[c];
    a[128] = (f16)(pr[0] - cr[0]);
    a[129] = (f16)(pr[1] - cr[1]);
    a[130] = (f16)(pr[2] - cr[2]);
    for (int c = 131; c < 160; c++) a[c] = (f16)0.f;
}

// Max-pool over valid neighbors (cnt >= 1, so no -inf rows survive).
__global__ void k_maxpool(const f16* __restrict__ h, const int* __restrict__ cnt,
                          f16* __restrict__ xout, int S, int C, int b)
{
    int g = blockIdx.x * blockDim.x + threadIdx.x;
    if (g >= S * C) return;
    int s = g / C, c = g % C;
    int cn = cnt[(size_t)b * S + s];
    float m = -3.0e38f;
    const f16* hr = h + (size_t)s * KNN * C + c;
    for (int k = 0; k < cn; k++) {
        float v = (float)hr[(size_t)k * C];
        if (v > m) m = v;
    }
    xout[((size_t)b * S + s) * C + c] = (f16)m;
}

// SA3 input: concat(x2[256], pos2[3]) -> [B*S2, 288]
__global__ void k_concat_sa3(const f16* __restrict__ x2, const float* __restrict__ pos2,
                             f16* __restrict__ A)
{
    int g = blockIdx.x * blockDim.x + threadIdx.x;
    if (g >= BB * SA2S) return;
    f16* a = A + (size_t)g * 288;
    const f16* xr = x2 + (size_t)g * 256;
    for (int c = 0; c < 256; c++) a[c] = xr[c];
    const float* pr = pos2 + (size_t)g * 3;
    a[256] = (f16)pr[0]; a[257] = (f16)pr[1]; a[258] = (f16)pr[2];
    for (int c = 259; c < 288; c++) a[c] = (f16)0.f;
}

// Global max over S2 points per cloud -> x3[B,1024]
__global__ void k_globalmax(const f16* __restrict__ h3, f16* __restrict__ x3)
{
    int g = blockIdx.x * blockDim.x + threadIdx.x;
    if (g >= BB * 1024) return;
    int b = g / 1024, c = g % 1024;
    float m = -3.0e38f;
    for (int s = 0; s < SA2S; s++) {
        float v = (float)h3[((size_t)b * SA2S + s) * 1024 + c];
        if (v > m) m = v;
    }
    x3[g] = (f16)m;
}

// FP3 input: knn k=1 from the single global point == broadcast x3.
__global__ void k_concat_fp3(const f16* __restrict__ x3, const f16* __restrict__ x2,
                             f16* __restrict__ A)
{
    int g = blockIdx.x * blockDim.x + threadIdx.x;
    if (g >= BB * SA2S) return;
    int b = g / SA2S;
    f16* a = A + (size_t)g * 1280;
    const f16* x3r = x3 + (size_t)b * 1024;
    for (int c = 0; c < 1024; c++) a[c] = x3r[c];
    const f16* x2r = x2 + (size_t)g * 256;
    for (int c = 0; c < 256; c++) a[1024 + c] = x2r[c];
}

// kNN (k=3) inverse-d^2 interpolation + skip concat -> padded f16 A.
// Strict '<' keeps the lower index on ties (matches lax.top_k).
__global__ void k_interp_concat(const f16* __restrict__ hsrc, const float* __restrict__ psrc,
                                int Ssrc, int Csrc,
                                const float* __restrict__ ptgt, int T,
                                const f16* __restrict__ skip16, const float* __restrict__ skip32,
                                int Cskip, f16* __restrict__ A, int CinP)
{
    int g = blockIdx.x * blockDim.x + threadIdx.x;
    if (g >= BB * T) return;
    int b = g / T;
    const float* pt = ptgt + (size_t)g * 3;
    const float* ps = psrc + (size_t)b * Ssrc * 3;
    float tx = pt[0], ty = pt[1], tz = pt[2];
    float d0 = 3.0e38f, d1 = 3.0e38f, d2v = 3.0e38f;
    int i0 = 0, i1 = 0, i2 = 0;
    for (int j = 0; j < Ssrc; j++) {
        float dx = ps[j * 3] - tx, dy = ps[j * 3 + 1] - ty, dz = ps[j * 3 + 2] - tz;
        float d = dx * dx + dy * dy + dz * dz;
        if (d < d0)      { d2v = d1; i2 = i1; d1 = d0; i1 = i0; d0 = d; i0 = j; }
        else if (d < d1) { d2v = d1; i2 = i1; d1 = d;  i1 = j; }
        else if (d < d2v){ d2v = d;  i2 = j; }
    }
    float w0 = 1.f / fmaxf(d0, 1e-16f);
    float w1 = 1.f / fmaxf(d1, 1e-16f);
    float w2 = 1.f / fmaxf(d2v, 1e-16f);
    float ws = w0 + w1 + w2;
    const f16* h0 = hsrc + ((size_t)b * Ssrc + i0) * Csrc;
    const f16* h1 = hsrc + ((size_t)b * Ssrc + i1) * Csrc;
    const f16* h2 = hsrc + ((size_t)b * Ssrc + i2) * Csrc;
    f16* a = A + (size_t)g * CinP;
    for (int c = 0; c < Csrc; c++)
        a[c] = (f16)((w0 * (float)h0[c] + w1 * (float)h1[c] + w2 * (float)h2[c]) / ws);
    for (int c = 0; c < Cskip; c++)
        a[Csrc + c] = skip16 ? skip16[(size_t)g * Cskip + c]
                             : (f16)skip32[(size_t)g * Cskip + c];
    for (int c = Csrc + Cskip; c < CinP; c++) a[c] = (f16)0.f;
}

// Extract 3 real output channels from the Np=16 padded f32 lin2 result.
__global__ void k_extract(const float* __restrict__ lin2out, float* __restrict__ out)
{
    int g = blockIdx.x * blockDim.x + threadIdx.x;
    if (g >= BB * NPTS * 3) return;
    int row = g / 3, c = g % 3;
    out[g] = lin2out[(size_t)row * 16 + c];
}

// =====================================================================
// Host orchestration
// =====================================================================
extern "C" void kernel_launch(void* const* d_in, const int* in_sizes, int n_in,
                              void* d_out, int out_size, void* d_ws, size_t ws_size,
                              hipStream_t stream)
{
    (void)in_sizes; (void)n_in; (void)out_size; (void)ws_size;

    const float* x_in  = (const float*)d_in[0];   // [8,2048,9]
    const float* pos   = (const float*)d_in[1];   // [8,2048,3]

    // Layer table: {d_in W index, cin, cout, cinP, coutP}
    static const int LW[18]     = {2,4,6, 8,10,12, 14,16,18, 20,22, 24,26, 28,30,32, 34,36};
    static const int LCIN[18]   = {12,64,64, 131,128,128, 259,256,512, 1280,256, 384,256, 137,128,128, 128,128};
    static const int LCOUT[18]  = {64,64,128, 128,128,256, 256,512,1024, 256,256, 256,128, 128,128,128, 128,3};
    static const int LCINP[18]  = {32,64,64, 160,128,128, 288,256,512, 1280,256, 384,256, 160,128,128, 128,128};
    static const int LCOUTP[18] = {64,64,128, 128,128,256, 256,512,1024, 256,256, 256,128, 128,128,128, 128,16};

    // ---- workspace bump allocator (every byte read is rewritten each call) ----
    char* wp = (char*)d_ws;
    auto alloc = [&](size_t bytes) -> void* {
        void* p = (void*)wp;
        wp += (bytes + 255) & ~(size_t)255;
        return p;
    };

    f16*   Wt[18];
    float* Bp[18];
    for (int l = 0; l < 18; l++) {
        Wt[l] = (f16*)alloc((size_t)LCOUTP[l] * LCINP[l] * sizeof(f16));
        Bp[l] = (float*)alloc((size_t)LCOUTP[l] * sizeof(float));
    }
    int*   idx1 = (int*)alloc((size_t)BB * SA1S * sizeof(int));
    int*   idx2 = (int*)alloc((size_t)BB * SA2S * sizeof(int));
    float* pos1 = (float*)alloc((size_t)BB * SA1S * 3 * sizeof(float));
    float* pos2 = (float*)alloc((size_t)BB * SA2S * 3 * sizeof(float));
    int*   nbr1 = (int*)alloc((size_t)BB * SA1S * KNN * sizeof(int));
    int*   cnt1 = (int*)alloc((size_t)BB * SA1S * sizeof(int));
    int*   nbr2 = (int*)alloc((size_t)BB * SA2S * KNN * sizeof(int));
    int*   cnt2 = (int*)alloc((size_t)BB * SA2S * sizeof(int));
    f16*   x1   = (f16*)alloc((size_t)BB * SA1S * 128 * sizeof(f16));
    f16*   x2   = (f16*)alloc((size_t)BB * SA2S * 256 * sizeof(f16));
    f16*   x3   = (f16*)alloc((size_t)BB * 1024 * sizeof(f16));
    f16*   hfp3 = (f16*)alloc((size_t)BB * SA2S * 256 * sizeof(f16));
    f16*   hfp2 = (f16*)alloc((size_t)BB * SA1S * 128 * sizeof(f16));
    float* l2o  = (float*)alloc((size_t)BB * NPTS * 16 * sizeof(float));
    f16*   bufA = (f16*)alloc((size_t)SA1S * KNN * 64 * sizeof(f16));   // 8 MB
    f16*   bufB = (f16*)alloc((size_t)SA1S * KNN * 128 * sizeof(f16));  // 16 MB

    auto gemm = [&](const f16* A, int l, void* out, int M, int relu, int of32) {
        int mtiles = M / 16;
        if (LCOUTP[l] % 64 == 0) {
            dim3 grd((mtiles + 3) / 4, LCOUTP[l] / 64);
            k_gemm<4><<<grd, dim3(128), 0, stream>>>(A, Wt[l], Bp[l], out,
                                                     M, LCINP[l], LCOUTP[l], relu, of32);
        } else {
            dim3 grd((mtiles + 3) / 4, LCOUTP[l] / 16);
            k_gemm<1><<<grd, dim3(128), 0, stream>>>(A, Wt[l], Bp[l], out,
                                                     M, LCINP[l], LCOUTP[l], relu, of32);
        }
    };

    // ---- weight prep ----
    for (int l = 0; l < 18; l++) {
        int total = LCOUTP[l] * LCINP[l];
        k_prep_weights<<<(total + 255) / 256, 256, 0, stream>>>(
            (const float*)d_in[LW[l]], (const float*)d_in[LW[l] + 1],
            Wt[l], Bp[l], LCIN[l], LCOUT[l], LCINP[l], LCOUTP[l]);
    }

    // ---- SA1 ----
    k_fps<<<BB, 1024, 0, stream>>>(pos, NPTS, SA1S, idx1);
    k_ballquery<<<(BB * SA1S + 255) / 256, 256, 0, stream>>>(
        pos, NPTS, idx1, SA1S, 0.04f, pos1, nbr1, cnt1);
    for (int b = 0; b < BB; b++) {
        k_gather_sa1<<<(SA1S * KNN + 255) / 256, 256, 0, stream>>>(x_in, pos, pos1, nbr1, bufA, b);
        gemm(bufA, 0, bufB, SA1S * KNN, 1, 0);
        gemm(bufB, 1, bufA, SA1S * KNN, 1, 0);
        gemm(bufA, 2, bufB, SA1S * KNN, 1, 0);
        k_maxpool<<<(SA1S * 128 + 255) / 256, 256, 0, stream>>>(bufB, cnt1, x1, SA1S, 128, b);
    }

    // ---- SA2 ----
    k_fps<<<BB, 1024, 0, stream>>>(pos1, SA1S, SA2S, idx2);
    k_ballquery<<<(BB * SA2S + 255) / 256, 256, 0, stream>>>(
        pos1, SA1S, idx2, SA2S, 0.16f, pos2, nbr2, cnt2);
    for (int b = 0; b < BB; b++) {
        k_gather_sa2<<<(SA2S * KNN + 255) / 256, 256, 0, stream>>>(x1, pos1, pos2, nbr2, bufA, b);
        gemm(bufA, 3, bufB, SA2S * KNN, 1, 0);
        gemm(bufB, 4, bufA, SA2S * KNN, 1, 0);
        gemm(bufA, 5, bufB, SA2S * KNN, 1, 0);
        k_maxpool<<<(SA2S * 256 + 255) / 256, 256, 0, stream>>>(bufB, cnt2, x2, SA2S, 256, b);
    }

    // ---- Global SA3 ----
    k_concat_sa3<<<(BB * SA2S + 255) / 256, 256, 0, stream>>>(x2, pos2, bufA);
    gemm(bufA, 6, bufB, BB * SA2S, 1, 0);
    gemm(bufB, 7, bufA, BB * SA2S, 1, 0);
    gemm(bufA, 8, bufB, BB * SA2S, 1, 0);
    k_globalmax<<<(BB * 1024 + 255) / 256, 256, 0, stream>>>(bufB, x3);

    // ---- FP3 (k=1 interp == broadcast) ----
    k_concat_fp3<<<(BB * SA2S + 255) / 256, 256, 0, stream>>>(x3, x2, bufA);
    gemm(bufA, 9, bufB, BB * SA2S, 1, 0);
    gemm(bufB, 10, hfp3, BB * SA2S, 1, 0);

    // ---- FP2 (k=3 interp pos2 -> pos1, skip x1) ----
    k_interp_concat<<<(BB * SA1S + 255) / 256, 256, 0, stream>>>(
        hfp3, pos2, SA2S, 256, pos1, SA1S, x1, (const float*)nullptr, 128, bufA, 384);
    gemm(bufA, 11, bufB, BB * SA1S, 1, 0);
    gemm(bufB, 12, hfp2, BB * SA1S, 1, 0);

    // ---- FP1 (k=3 interp pos1 -> pos, skip x) ----
    k_interp_concat<<<(BB * NPTS + 255) / 256, 256, 0, stream>>>(
        hfp2, pos1, SA1S, 128, pos, NPTS, (const f16*)nullptr, x_in, NFEAT, bufA, 160);
    gemm(bufA, 13, bufB, BB * NPTS, 1, 0);
    gemm(bufB, 14, bufA, BB * NPTS, 1, 0);
    gemm(bufA, 15, bufB, BB * NPTS, 1, 0);

    // ---- head: lin1 (ReLU) + lin2 (f32 out, Np padded to 16) ----
    gemm(bufB, 16, bufA, BB * NPTS, 1, 0);
    gemm(bufA, 17, l2o, BB * NPTS, 0, 1);
    k_extract<<<(BB * NPTS * 3 + 255) / 256, 256, 0, stream>>>(l2o, (float*)d_out);
}
